// SiamesePXT_30605936951337
// MI455X (gfx1250) — matrix operand order repaired and checked by reading, not compiled, not run on hardware
//
#include <hip/hip_runtime.h>
#include <hip/hip_bf16.h>

typedef __attribute__((ext_vector_type(16))) _Float16 v16h;
typedef __attribute__((ext_vector_type(8)))  float    v8f;

#define BN_EPS 1e-5f

__device__ __forceinline__ int iclamp(int v, int lo, int hi) {
    return v < lo ? lo : (v > hi ? hi : v);
}

// ---------------------------------------------------------------------------
// Depthwise cross-correlation (VALID) via per-kernel-row Toeplitz WMMA.
// One wave computes one 16x16 output tile of one (batch,chan) plane:
//   out[r,n] = sum_i sum_j s[p0+r+i, q0+n+j] * k[i,j]
//            = sum_i ( S_i (16 x L) * T_i (L x 16) ),  T_i[j',n] = k[i, j'-n]
// Loads are unconditional (clamped addresses); out-of-range elements are
// zeroed with multiply-by-mask so the backend cannot sink loads under
// divergent conditions.  Interior A fills use aligned float4 loads.
// ---------------------------------------------------------------------------
__global__ __launch_bounds__(32)
void corr_wmma_kernel(const float* __restrict__ s, const float* __restrict__ k,
                      float* __restrict__ out, int Hs, int Ho, int Wout, int tilesW)
{
    const int lane  = threadIdx.x & 31;
    const int plane = blockIdx.y;
    const int p0    = (blockIdx.x / tilesW) * 16;
    const int q0    = (blockIdx.x % tilesW) * 16;
    const int m     = lane & 15;           // A row / D col index
    const int hi    = (lane >> 4) & 1;     // lane half
    const int klo   = hi ? 8 : 0;
    const int kbB   = hi ? 8 : 0;
    const int L     = Ho + 15;             // Toeplitz K extent per kernel row

    const float* sp = s + (size_t)plane * Hs * Hs;
    const float* kp = k + (size_t)plane * Ho * Ho;

    v8f acc = {};
    for (int i = 0; i < Ho; ++i) {
        const int   row   = p0 + m + i;
        const float rmask = (row < Hs) ? 1.0f : 0.0f;   // per-lane
        const bool rowsOk = (p0 + 15 + i) < Hs;         // uniform
        const float* srow = sp + (size_t)iclamp(row, 0, Hs - 1) * Hs;
        const float* krow = kp + (size_t)i * Ho;

        for (int cb = 0; cb < L; cb += 32) {
            float ta[16], tb[16];

            // ---- A: cols q0+cb .. q0+cb+31 of row ----
            if (rowsOk && (q0 + cb + 31) < Hs) {        // uniform fast path
                const float4* p4 =
                    reinterpret_cast<const float4*>(srow + q0 + cb + klo);
                float4 x0 = p4[0], x1 = p4[1], x2 = p4[4], x3 = p4[5];
                ta[0]=x0.x;  ta[1]=x0.y;  ta[2]=x0.z;  ta[3]=x0.w;
                ta[4]=x1.x;  ta[5]=x1.y;  ta[6]=x1.z;  ta[7]=x1.w;
                ta[8]=x2.x;  ta[9]=x2.y;  ta[10]=x2.z; ta[11]=x2.w;
                ta[12]=x3.x; ta[13]=x3.y; ta[14]=x3.z; ta[15]=x3.w;
            } else {
#pragma unroll
                for (int e = 0; e < 16; ++e) {
                    int col = q0 + cb + klo + e + ((e >= 8) ? 8 : 0);
                    ta[e] = srow[iclamp(col, 0, Hs - 1)];   // always in-bounds
                }
#pragma unroll
                for (int e = 0; e < 16; ++e) {
                    int col = q0 + cb + klo + e + ((e >= 8) ? 8 : 0);
                    ta[e] *= (col < Hs) ? rmask : 0.0f;
                }
            }

            // ---- B: Toeplitz taps (cb+kbB-m) + {0..7, 16..23} of kernel row i ----
            if (cb >= 15 && (cb + 31) < Ho) {           // uniform fast path
                const float* pB = krow + (cb + kbB - m);
#pragma unroll
                for (int e = 0; e < 16; ++e) tb[e] = pB[e + ((e >= 8) ? 8 : 0)];
            } else {
#pragma unroll
                for (int e = 0; e < 16; ++e) {
                    int tap = cb + kbB + e + ((e >= 8) ? 8 : 0) - m;
                    tb[e] = krow[iclamp(tap, 0, Ho - 1)];
                }
#pragma unroll
                for (int e = 0; e < 16; ++e) {
                    int tap = cb + kbB + e + ((e >= 8) ? 8 : 0) - m;
                    tb[e] *= (tap >= 0 && tap < Ho) ? 1.0f : 0.0f;
                }
            }

            v16h a, b;
#pragma unroll
            for (int e = 0; e < 16; ++e) {
                a[e] = (_Float16)ta[e];
                b[e] = (_Float16)tb[e];
            }
            acc = __builtin_amdgcn_wmma_f32_16x16x32_f16(
                      false, a, false, b, (short)0, acc, false, false);
        }
    }

    float* op = out + (size_t)plane * Wout * Wout;
#pragma unroll
    for (int v = 0; v < 8; ++v) {
        int pp = p0 + v + (hi ? 8 : 0);
        int qq = q0 + m;
        if (pp < Wout && qq < Wout) op[(size_t)pp * Wout + qq] = acc[v];
    }
}

// ---------------------------------------------------------------------------
// Conv3x3 (pad=1) + BN(eval) + ReLU via implicit-GEMM WMMA.
// One wave computes a 16(pixels along W) x 16(out-channels) tile for one
// (batch, output row).  kh outer (uniform row-guard skip); K = Cin*3.
// Same unconditional-load + multiply-mask structure as the corr kernel.
// ---------------------------------------------------------------------------
__global__ __launch_bounds__(32)
void conv3x3_bn_relu_wmma(const float* __restrict__ x,  const float* __restrict__ wt,
                          const float* __restrict__ bias, const float* __restrict__ g,
                          const float* __restrict__ be,   const float* __restrict__ mean,
                          const float* __restrict__ var,
                          float* __restrict__ y,
                          int Cin, int Cout, int H, int W, int tilesW)
{
    const int lane  = threadIdx.x & 31;
    const int w0    = (blockIdx.x % tilesW) * 16;
    const int h     =  blockIdx.x / tilesW;
    const int batch =  blockIdx.y;
    const int n0    =  blockIdx.z * 16;
    const int m     = lane & 15;
    const int hi    = (lane >> 4) & 1;
    const int klo   = hi ? 8 : 0;
    const int kbB   = hi ? 8 : 0;
    const int HW    = H * W;
    const int K3    = Cin * 3;

    const float* xb = x + (size_t)batch * Cin * HW;

    const int  co    = n0 + m;
    const bool coOk  = co < Cout;
    const float cmask = coOk ? 1.0f : 0.0f;
    const int  coC   = coOk ? co : (Cout - 1);
    const bool nOk   = (n0 + 15) < Cout;              // uniform

    v8f acc = {};
    for (int kh = 0; kh < 3; ++kh) {
        const int hh = h + kh - 1;
        if (hh < 0 || hh >= H) continue;              // uniform skip
        const float* xrow  = xb + (size_t)hh * W;     // + ci*HW + ww
        const float* wbase = wt + (size_t)coC * Cin * 9 + kh * 3;  // + ci*9 + kw

        for (int kbase = 0; kbase < K3; kbase += 32) {
            float ta[16], tb[16];

            // ---- A fill ----
            if ((kbase + 31) < K3 && w0 >= 1 && (w0 + 16) < W) {   // uniform fast
#pragma unroll
                for (int e = 0; e < 16; ++e) {
                    int kk = kbase + klo + e + ((e >= 8) ? 8 : 0);
                    int ci = kk / 3;
                    int kw = kk - 3 * ci;
                    ta[e] = xrow[ci * HW + (w0 + m + kw - 1)];
                }
            } else {
#pragma unroll
                for (int e = 0; e < 16; ++e) {
                    int kk = kbase + klo + e + ((e >= 8) ? 8 : 0);
                    int ci = kk / 3;
                    int kw = kk - 3 * ci;
                    int ww = w0 + m + kw - 1;
                    ta[e] = xrow[iclamp(ci, 0, Cin - 1) * HW + iclamp(ww, 0, W - 1)];
                }
#pragma unroll
                for (int e = 0; e < 16; ++e) {
                    int kk = kbase + klo + e + ((e >= 8) ? 8 : 0);
                    int ci = kk / 3;
                    int kw = kk - 3 * ci;
                    int ww = w0 + m + kw - 1;
                    ta[e] *= ((kk < K3) && (ww >= 0) && (ww < W)) ? 1.0f : 0.0f;
                }
            }

            // ---- B fill ----
            if ((kbase + 31) < K3 && nOk) {                         // uniform fast
#pragma unroll
                for (int e = 0; e < 16; ++e) {
                    int kk = kbase + kbB + e + ((e >= 8) ? 8 : 0);
                    int ci = kk / 3;
                    int kw = kk - 3 * ci;
                    tb[e] = wbase[ci * 9 + kw];
                }
            } else {
#pragma unroll
                for (int e = 0; e < 16; ++e) {
                    int kc = iclamp(kbase + kbB + e + ((e >= 8) ? 8 : 0), 0, K3 - 1);
                    int ci = kc / 3;
                    int kw = kc - 3 * ci;
                    tb[e] = wbase[ci * 9 + kw];
                }
#pragma unroll
                for (int e = 0; e < 16; ++e) {
                    int kk = kbase + kbB + e + ((e >= 8) ? 8 : 0);
                    tb[e] *= (kk < K3) ? cmask : 0.0f;
                }
            }

            v16h a, b;
#pragma unroll
            for (int e = 0; e < 16; ++e) {
                a[e] = (_Float16)ta[e];
                b[e] = (_Float16)tb[e];
            }
            acc = __builtin_amdgcn_wmma_f32_16x16x32_f16(
                      false, a, false, b, (short)0, acc, false, false);
        }
    }

    if (coOk) {
        float inv   = rsqrtf(var[co] + BN_EPS);
        float scale = g[co] * inv;
        float shift = (bias[co] - mean[co]) * scale + be[co];
        float* yb = y + (((size_t)batch * Cout + co) * H + h) * W;
#pragma unroll
        for (int v = 0; v < 8; ++v) {
            int ww = w0 + v + (hi ? 8 : 0);
            if (ww < W) {
                float val = acc[v] * scale + shift;
                yb[ww] = val > 0.0f ? val : 0.0f;
            }
        }
    }
}

// ---------------------------------------------------------------------------
// fout[b,oh,ow] += bilinear_upsample(cin[b], Hi -> Ho) (half-pixel, clamped)
// ---------------------------------------------------------------------------
__global__ void up_add_kernel(const float* __restrict__ cin, float* __restrict__ fout,
                              int Hi, int Ho, int nB)
{
    int idx   = blockIdx.x * blockDim.x + threadIdx.x;
    int total = nB * Ho * Ho;
    if (idx >= total) return;
    int ow = idx % Ho;
    int t  = idx / Ho;
    int oh = t % Ho;
    int b  = t / Ho;

    float scale = (float)Hi / (float)Ho;
    float sy = (oh + 0.5f) * scale - 0.5f;
    float sx = (ow + 0.5f) * scale - 0.5f;
    sy = fminf(fmaxf(sy, 0.0f), (float)(Hi - 1));
    sx = fminf(fmaxf(sx, 0.0f), (float)(Hi - 1));
    int y0 = (int)sy;
    int x0 = (int)sx;
    int y1 = (y0 + 1 < Hi) ? y0 + 1 : Hi - 1;
    int x1 = (x0 + 1 < Hi) ? x0 + 1 : Hi - 1;
    float fy = sy - (float)y0;
    float fx = sx - (float)x0;

    const float* cp = cin + (size_t)b * Hi * Hi;
    float v = (1.0f - fy) * ((1.0f - fx) * cp[y0 * Hi + x0] + fx * cp[y0 * Hi + x1])
            +         fy  * ((1.0f - fx) * cp[y1 * Hi + x0] + fx * cp[y1 * Hi + x1]);
    fout[idx] += v;
}

// ---------------------------------------------------------------------------
// Host side
// ---------------------------------------------------------------------------
extern "C" void kernel_launch(void* const* d_in, const int* in_sizes, int n_in,
                              void* d_out, int out_size, void* d_ws, size_t ws_size,
                              hipStream_t stream)
{
    (void)n_in; (void)out_size; (void)ws_size;

    static const int chains[4][5] = {
        {24, 16, 16, 16, 1}, {32, 16, 16, 16, 1},
        {96, 48, 24, 16, 1}, {320, 160, 80, 40, 1}};
    static const int HsA[4] = {128, 64, 32, 16};
    static const int HoA[4] = {64, 32, 16, 8};
    static const int WoA[4] = {65, 33, 17, 9};
    const int Bn = 32;

    // Input order: setup_inputs() dict order interleaves obj/s; auto-detect.
    const float* obj[4];
    const float* srch[4];
    bool interleaved = (in_sizes[1] == 32 * 24 * 128 * 128);   // s_b1 right after obj_b1
    for (int br = 0; br < 4; ++br) {
        if (interleaved) { obj[br] = (const float*)d_in[2 * br];
                           srch[br] = (const float*)d_in[2 * br + 1]; }
        else             { obj[br] = (const float*)d_in[br];
                           srch[br] = (const float*)d_in[4 + br]; }
    }

    // params: 4 branches x 4 layers x 6 arrays, order auto-detected per layer.
    const float *Pw[4][4], *Pb[4][4], *Pg[4][4], *Pbe[4][4], *Pm[4][4], *Pv[4][4];
    int idx = 8;
    for (int br = 0; br < 4; ++br) {
        for (int l = 0; l < 4; ++l) {
            int cin = chains[br][l], cout = chains[br][l + 1];
            int wsz = cout * cin * 9;
            if (in_sizes[idx] == wsz) {   // insertion order: w,b,g,be,mean,var
                Pw[br][l]  = (const float*)d_in[idx + 0];
                Pb[br][l]  = (const float*)d_in[idx + 1];
                Pg[br][l]  = (const float*)d_in[idx + 2];
                Pbe[br][l] = (const float*)d_in[idx + 3];
                Pm[br][l]  = (const float*)d_in[idx + 4];
                Pv[br][l]  = (const float*)d_in[idx + 5];
            } else {                      // pytree-sorted: b,be,g,mean,var,w
                Pb[br][l]  = (const float*)d_in[idx + 0];
                Pbe[br][l] = (const float*)d_in[idx + 1];
                Pg[br][l]  = (const float*)d_in[idx + 2];
                Pm[br][l]  = (const float*)d_in[idx + 3];
                Pv[br][l]  = (const float*)d_in[idx + 4];
                Pw[br][l]  = (const float*)d_in[idx + 5];
            }
            idx += 6;
        }
    }

    // Workspace layout (floats)
    float* W = (float*)d_ws;
    const size_t OFF_C[4] = {0, 3244800, 4359936, 5247744};   // corr buffers
    const size_t OFF_T0 = 6077184;                            // ping
    const size_t OFF_T1 = 8240384;                            // pong
    const size_t OFF_S4 = 10403584;                           // 32*9*9
    const size_t OFF_S3 = 10406176;                           // 32*17*17
    const size_t OFF_S2 = 10415424;                           // 32*33*33
    float* corrbuf[4] = {W + OFF_C[0], W + OFF_C[1], W + OFF_C[2], W + OFF_C[3]};
    float* sc2 = W + OFF_S2;
    float* sc3 = W + OFF_S3;
    float* sc4 = W + OFF_S4;

    for (int br = 0; br < 4; ++br) {
        const int Hs = HsA[br], Ho = HoA[br], Wo = WoA[br];
        const int planes = Bn * chains[br][0];
        const int tiles  = (Wo + 15) / 16;

        dim3 gc(tiles * tiles, planes, 1);
        corr_wmma_kernel<<<gc, 32, 0, stream>>>(srch[br], obj[br], corrbuf[br],
                                                Hs, Ho, Wo, tiles);

        const float* xin = corrbuf[br];
        float* bufs[2] = {W + OFF_T0, W + OFF_T1};
        for (int l = 0; l < 4; ++l) {
            int cin = chains[br][l], cout = chains[br][l + 1];
            float* yout;
            if (l == 3) {
                if (br == 0)      yout = (float*)d_out;  // sc_b1 -> final buffer
                else if (br == 1) yout = sc2;
                else if (br == 2) yout = sc3;
                else              yout = sc4;
            } else {
                yout = bufs[l & 1];
            }
            int tilesW = (Wo + 15) / 16;
            dim3 gv(tilesW * Wo, Bn, (cout + 15) / 16);
            conv3x3_bn_relu_wmma<<<gv, 32, 0, stream>>>(
                xin, Pw[br][l], Pb[br][l], Pg[br][l], Pbe[br][l], Pm[br][l],
                Pv[br][l], yout, cin, cout, Wo, Wo, tilesW);
            xin = yout;
        }
    }

    // Pyramid combine: sc3 += up(sc4); sc2 += up(sc3); out += up(sc2)
    {
        int total = Bn * 17 * 17;
        up_add_kernel<<<(total + 255) / 256, 256, 0, stream>>>(sc4, sc3, 9, 17, Bn);
        total = Bn * 33 * 33;
        up_add_kernel<<<(total + 255) / 256, 256, 0, stream>>>(sc3, sc2, 17, 33, Bn);
        total = Bn * 65 * 65;
        up_add_kernel<<<(total + 255) / 256, 256, 0, stream>>>(sc2, (float*)d_out,
                                                               33, 65, Bn);
    }
}
